// GatedGCNLayer_10943576670413
// MI455X (gfx1250) — compile-verified, matrix operand order checked by reading, not run on hardware
//
#include <hip/hip_runtime.h>

#define N_NODES 50000
#define N_EDGES 800000
#define DDIM 128
#define BN_EPS 1e-5f

typedef _Float16 v16h __attribute__((ext_vector_type(16)));
typedef _Float16 v8h  __attribute__((ext_vector_type(8)));
typedef _Float16 v4h  __attribute__((ext_vector_type(4)));
typedef float    v8f  __attribute__((ext_vector_type(8)));
typedef float    v4f  __attribute__((ext_vector_type(4)));

// Load a WMMA A/B fragment (16-bit, 16x32 layout) for this lane from a
// row-major f16 buffer. p points at (row = lane%16, k = kbase + (lane/16)*8).
// VGPRs 0-3 hold K = khalf..khalf+7 (packed pairs), VGPRs 4-7 hold K+16.
__device__ __forceinline__ v16h load_frag(const _Float16* p) {
  v8h lo = *(const v8h*)p;
  v8h hi = *(const v8h*)(p + 16);
  v16h r;
#pragma unroll
  for (int i = 0; i < 8; ++i) { r[i] = lo[i]; r[i + 8] = hi[i]; }
  return r;
}

__device__ __forceinline__ v8f wmma_f16(const v16h& a, const v16h& b, const v8f& c) {
  return __builtin_amdgcn_wmma_f32_16x16x32_f16(false, a, false, b,
                                                (short)0, c, false, false);
}

// ---------------------------------------------------------------- prep ----
__global__ void __launch_bounds__(256) prep_kernel(
    const float* __restrict__ WA, const float* __restrict__ WB,
    const float* __restrict__ WC, const float* __restrict__ WD,
    const float* __restrict__ WE, _Float16* __restrict__ w16,
    float* __restrict__ stats) {
  int idx = blockIdx.x * 256 + threadIdx.x;
  if (idx < 5 * DDIM * DDIM) {
    int m = idx >> 14;
    int off = idx & (DDIM * DDIM - 1);
    const float* W = (m == 0) ? WA : (m == 1) ? WB : (m == 2) ? WC
                   : (m == 3) ? WD : WE;
    w16[idx] = (_Float16)W[off];
  }
  if (idx < 2 * DDIM) stats[idx] = 0.0f;
}

// ----------------------------------------------------- node projections ----
// Per wave: 16x128 strip of one of the 4 projections. Preload all B frags,
// back-to-back WMMAs, fully unrolled so loads pipeline under WMMAs.
template <bool FULL>
__device__ __forceinline__ void node_gemm_body(
    const v16h* af, const _Float16* __restrict__ W, float* __restrict__ outp,
    int rowbase, int lr, int half) {
#pragma unroll
  for (int nt = 0; nt < 8; ++nt) {
    v16h bf[4];
#pragma unroll
    for (int kc = 0; kc < 4; ++kc)
      bf[kc] = load_frag(W + (nt * 16 + lr) * DDIM + kc * 32 + half * 8);
    v8f acc = {};
#pragma unroll
    for (int kc = 0; kc < 4; ++kc) acc = wmma_f16(af[kc], bf[kc], acc);
    int col = nt * 16 + lr;
#pragma unroll
    for (int v = 0; v < 8; ++v) {
      int gr = rowbase + v + half * 8;
      if (FULL || gr < N_NODES) outp[gr * DDIM + col] = acc[v];
    }
  }
}

__global__ void __launch_bounds__(256) node_gemm_kernel(
    const float* __restrict__ h, const _Float16* __restrict__ w16,
    float* __restrict__ Ah, float* __restrict__ Bh, float* __restrict__ Ch,
    float* __restrict__ Dh) {
  __shared__ __align__(16) _Float16 hs[32 * 136];  // 68 dwords/row: bank-safe
  int t = threadIdx.x;
  int row0 = blockIdx.x * 32;
  bool full = (row0 + 32 <= N_NODES);

  if (full) {
#pragma unroll
    for (int idx = t; idx < 32 * 32; idx += 256) {      // float4 granules
      int r = idx >> 5, c4 = idx & 31;
      v4f v = ((const v4f*)h)[(size_t)(row0 + r) * 32 + c4];
      v4h hv;
#pragma unroll
      for (int i = 0; i < 4; ++i) hv[i] = (_Float16)v[i];
      *(v4h*)&hs[r * 136 + c4 * 4] = hv;
    }
  } else {
    for (int idx = t; idx < 32 * DDIM; idx += 256) {
      int r = idx >> 7, c = idx & 127;
      int gr = row0 + r;
      float v = (gr < N_NODES) ? h[gr * DDIM + c] : 0.0f;
      hs[r * 136 + c] = (_Float16)v;
    }
  }
  __syncthreads();

  int w = t >> 5, lane = t & 31;
  int m = w & 3, rt = w >> 2;
  int lr = lane & 15, half = lane >> 4;
  int arow = rt * 16 + lr;

  v16h af[4];
#pragma unroll
  for (int kc = 0; kc < 4; ++kc)
    af[kc] = load_frag(&hs[arow * 136 + kc * 32 + half * 8]);

  const _Float16* W = w16 + m * DDIM * DDIM;
  float* outp = (m == 0) ? Ah : (m == 1) ? Bh : (m == 2) ? Ch : Dh;
  int rowbase = row0 + rt * 16;

  if (full) node_gemm_body<true>(af, W, outp, rowbase, lr, half);
  else      node_gemm_body<false>(af, W, outp, rowbase, lr, half);
}

// ------------------------------------------------- fused edge GEMM+gate ----
// Block: 64 edges (E = 12500 * 64 exactly, no tail). 8 waves = 4 edge-tiles
// x 2 column halves. e_new = e @ WE^T via WMMA, then gate/message/scatter.
__global__ void __launch_bounds__(256) edge_kernel(
    const float* __restrict__ e, const _Float16* __restrict__ we16,
    const float* __restrict__ Ah, const float* __restrict__ Bh,
    const float* __restrict__ Ch, const int* __restrict__ src,
    const int* __restrict__ dst, float* __restrict__ hout,
    float* __restrict__ eout) {
  __shared__ __align__(16) _Float16 es[64 * 136];
  __shared__ int sidx[64];
  __shared__ int didx[64];
  int t = threadIdx.x;
  int e0 = blockIdx.x * 64;
  if (t < 64) { sidx[t] = src[e0 + t]; didx[t] = dst[e0 + t]; }
#pragma unroll
  for (int idx = t; idx < 64 * 32; idx += 256) {        // float4 granules
    int r = idx >> 5, c4 = idx & 31;
    v4f v = ((const v4f*)e)[(size_t)(e0 + r) * 32 + c4];
    v4h hv;
#pragma unroll
    for (int i = 0; i < 4; ++i) hv[i] = (_Float16)v[i];
    *(v4h*)&es[r * 136 + c4 * 4] = hv;
  }
  __syncthreads();

  int w = t >> 5, lane = t & 31;
  int et = w >> 1, nh = w & 1;
  int lr = lane & 15, half = lane >> 4;
  int arow = et * 16 + lr;

  v16h af[4];
#pragma unroll
  for (int kc = 0; kc < 4; ++kc)
    af[kc] = load_frag(&es[arow * 136 + kc * 32 + half * 8]);

#pragma unroll
  for (int j = 0; j < 4; ++j) {
    int nt = nh * 4 + j;
    v16h bf[4];
#pragma unroll
    for (int kc = 0; kc < 4; ++kc)
      bf[kc] = load_frag(we16 + (nt * 16 + lr) * DDIM + kc * 32 + half * 8);
    v8f acc = {};
#pragma unroll
    for (int kc = 0; kc < 4; ++kc) acc = wmma_f16(af[kc], bf[kc], acc);

    int col = nt * 16 + lr;
#pragma unroll
    for (int v = 0; v < 8; ++v) {
      int ml = et * 16 + v + half * 8;   // local edge (half-wave uniform)
      int s = sidx[ml], d = didx[ml];
      float ev = acc[v];
      float g = Ah[s * DDIM + col] + Bh[d * DDIM + col] + ev;
      g = 1.0f / (1.0f + __expf(-g));
      float msg = Ch[s * DDIM + col] * g;
      atomicAdd(&hout[d * DDIM + col], msg);         // L2-resident scatter
      eout[(size_t)(e0 + ml) * DDIM + col] = ev;     // e_new output
    }
  }
}

// ---------------------------------------------------------- batch norm ----
__global__ void __launch_bounds__(256) bn_stats_kernel(
    const float* __restrict__ hout, float* __restrict__ stats) {
  __shared__ float red[256];
  int t = threadIdx.x;
  int col = t & 127, rl = t >> 7;
  float s = 0.f, s2 = 0.f;
  for (int row = blockIdx.x * 2 + rl; row < N_NODES; row += gridDim.x * 2) {
    float v = hout[row * DDIM + col];
    s += v; s2 += v * v;
  }
  red[t] = s;
  __syncthreads();
  float ts = (t < 128) ? (red[t] + red[t + 128]) : 0.f;
  __syncthreads();
  red[t] = s2;
  __syncthreads();
  if (t < 128) {
    float ts2 = red[t] + red[t + 128];
    atomicAdd(&stats[col], ts);
    atomicAdd(&stats[DDIM + col], ts2);
  }
}

__global__ void __launch_bounds__(256) bn_norm_kernel(
    float* __restrict__ hout, const float* __restrict__ stats,
    const float* __restrict__ gamma, const float* __restrict__ beta) {
  __shared__ float smu[DDIM], ssc[DDIM], sbe[DDIM];
  int t = threadIdx.x;
  if (t < DDIM) {
    const float invN = 1.0f / (float)N_NODES;
    float mu = stats[t] * invN;
    float var = stats[DDIM + t] * invN - mu * mu;
    smu[t] = mu;
    ssc[t] = rsqrtf(var + BN_EPS) * gamma[t];
    sbe[t] = beta[t];
  }
  __syncthreads();
  int idx4 = blockIdx.x * 256 + t;                 // float4 granules
  if (idx4 >= N_NODES * 32) return;
  int c4 = (idx4 & 31) * 4;
  v4f x = ((const v4f*)hout)[idx4];
  v4f y;
#pragma unroll
  for (int i = 0; i < 4; ++i) {
    int col = c4 + i;
    float v = (x[i] - smu[col]) * ssc[col] + sbe[col];
    y[i] = v > 0.f ? v : 0.f;
  }
  ((v4f*)hout)[idx4] = y;
}

// -------------------------------------------------------------- launch ----
extern "C" void kernel_launch(void* const* d_in, const int* in_sizes, int n_in,
                              void* d_out, int out_size, void* d_ws, size_t ws_size,
                              hipStream_t stream) {
  const float* h     = (const float*)d_in[0];
  const float* e     = (const float*)d_in[1];
  const float* WA    = (const float*)d_in[2];
  const float* WB    = (const float*)d_in[3];
  const float* WC    = (const float*)d_in[4];
  const float* WD    = (const float*)d_in[5];
  const float* WE    = (const float*)d_in[6];
  const float* gamma = (const float*)d_in[7];
  const float* beta  = (const float*)d_in[8];
  const int*   src   = (const int*)d_in[9];
  const int*   dst   = (const int*)d_in[10];

  float* hout = (float*)d_out;                       // [N, 128]
  float* eout = hout + (size_t)N_NODES * DDIM;       // [E, 128]

  // workspace layout
  char* ws = (char*)d_ws;
  _Float16* w16 = (_Float16*)ws;                                   // 5*128*128 f16
  size_t off = (size_t)5 * DDIM * DDIM * sizeof(_Float16);         // 163840
  float* stats = (float*)(ws + off);                               // 256 f32
  off += 2 * DDIM * sizeof(float);                                 // 164864 (256B aligned)
  float* Ah = (float*)(ws + off);
  float* Bh = Ah + (size_t)N_NODES * DDIM;
  float* Ch = Bh + (size_t)N_NODES * DDIM;

  prep_kernel<<<(5 * DDIM * DDIM + 255) / 256, 256, 0, stream>>>(
      WA, WB, WC, WD, WE, w16, stats);

  node_gemm_kernel<<<(N_NODES + 31) / 32, 256, 0, stream>>>(
      h, w16, Ah, Bh, Ch, hout);

  edge_kernel<<<N_EDGES / 64, 256, 0, stream>>>(
      e, w16 + (size_t)4 * DDIM * DDIM, Ah, Bh, Ch, src, dst, hout, eout);

  bn_stats_kernel<<<512, 256, 0, stream>>>(hout, stats);

  bn_norm_kernel<<<(N_NODES * 32 + 255) / 256, 256, 0, stream>>>(
      hout, stats, gamma, beta);
}
